// MambaEEGModel_1932735283989
// MI455X (gfx1250) — matrix operand order repaired and checked by reading, not compiled, not run on hardware
//
#include <hip/hip_runtime.h>
#include <math.h>

// ---------------------------------------------------------------------------
// MambaEEG forward for MI455X (gfx1250, wave32, WMMA).
// Dense projections -> v_wmma_f32_16x16x32_f16 (f16 mul / f32 acc), with
// vectorized b128 staging and contiguous b128 fragment gathers.
// Attention / SSM scan / elementwise -> f32 VALU + LDS kernels.
// ---------------------------------------------------------------------------

typedef __attribute__((ext_vector_type(16))) _Float16 v16h;
typedef __attribute__((ext_vector_type(8)))  _Float16 v8h;
typedef __attribute__((ext_vector_type(8)))  float    v8f;
typedef __attribute__((ext_vector_type(4)))  float    v4f;

#define DB 128      // batch
#define DC 129      // channels
#define KC 160      // channels padded to multiple of 32
#define DL 200      // time steps
#define DM 320      // d_model
#define DROWS (DB*DL)   // 25600 rows, multiple of 64
#define DIN 640     // 2*DM
#define HM 20       // mamba heads
#define DPROJ 1364  // 2*640 + 2*32 + 20
#define CDIM 704    // 640 + 64
#define NHEADS 16
#define HDIM 20
#define QTB 13      // ceil(200/16)

__device__ __forceinline__ float sigm(float x) { return 1.0f / (1.0f + expf(-x)); }
__device__ __forceinline__ float gelu_exact(float v) {
  return 0.5f * v * (1.0f + erff(v * 0.70710678118f));
}

__device__ __forceinline__ float block_reduce_sum(float v, float* red) {
  const int tid = threadIdx.x;
  red[tid] = v; __syncthreads();
  for (int off = 128; off > 0; off >>= 1) {
    if (tid < off) red[tid] += red[tid + off];
    __syncthreads();
  }
  float r = red[0]; __syncthreads();
  return r;
}
__device__ __forceinline__ float block_reduce_max(float v, float* red) {
  const int tid = threadIdx.x;
  red[tid] = v; __syncthreads();
  for (int off = 128; off > 0; off >>= 1) {
    if (tid < off) red[tid] = fmaxf(red[tid], red[tid + off]);
    __syncthreads();
  }
  float r = red[0]; __syncthreads();
  return r;
}

// ---------------------------------------------------------------------------
// WMMA GEMM:  C[M,N] = act( A[M,K] * W[N,K]^T + bias + addsrc )
// Requirements (guaranteed by launcher): M % 64 == 0, K % 32 == 0, A/W rows
// 16-byte aligned (K % 4 == 0). N arbitrary (zero-padded rows, guarded store).
// Block = 256 thr = 8 waves, tile 64x64; each wave 16x32 via two WMMA.
// Per K-step: 2x global b128 -> pack v8h -> 1x ds b128 per thread (A and W),
// then 6x ds_load_b128 fragment gathers + 2x v_wmma per wave.
// act: 0=none 1=gelu 2=tanh
// ---------------------------------------------------------------------------
__global__ __launch_bounds__(256)
void gemm_wmma_kernel(const float* __restrict__ A, const float* __restrict__ W,
                      const float* __restrict__ bias, const float* __restrict__ addsrc,
                      float* __restrict__ C, int M, int N, int K, int act)
{
  // rows padded to 40 halves (80 B) to stagger LDS banks; 16B-aligned rows.
  __shared__ _Float16 As[64][40];   // As[m][k]
  __shared__ _Float16 Bs[64][40];   // Bs[n][k]  (W tile, row per output col)
  const int tid  = threadIdx.x;
  const int lane = tid & 31;
  const int wv   = tid >> 5;        // 0..7
  const int wm   = wv >> 1;         // 0..3 (row group of 16)
  const int wn   = wv & 1;          // 0..1 (col group of 32)
  const int mbase = blockIdx.y * 64;
  const int nbase = blockIdx.x * 64;

  v8f acc0 = {};
  v8f acc1 = {};

  const int srow = tid >> 2;        // 0..63
  const int scol = (tid & 3) * 8;   // 0,8,16,24
  const bool nok = (nbase + srow) < N;
  const int nk = K >> 5;

  const int mr = lane & 15;
  const int kh = lane >> 4;

  for (int kb = 0; kb < nk; ++kb) {
    const int k0 = kb * 32;
    // stage A tile (64x32): two b128 loads -> one b128 LDS store
    {
      const v4f* ap = (const v4f*)(A + (size_t)(mbase + srow) * K + k0 + scol);
      const v4f a0 = ap[0], a1 = ap[1];
      v8h h;
      #pragma unroll
      for (int j = 0; j < 4; ++j) { h[j] = (_Float16)a0[j]; h[j + 4] = (_Float16)a1[j]; }
      *(v8h*)&As[srow][scol] = h;
    }
    // stage W tile (64 rows x 32 k): zero rows beyond N
    {
      v8h h = {};
      if (nok) {
        const v4f* wp = (const v4f*)(W + (size_t)(nbase + srow) * K + k0 + scol);
        const v4f w0 = wp[0], w1 = wp[1];
        #pragma unroll
        for (int j = 0; j < 4; ++j) { h[j] = (_Float16)w0[j]; h[j + 4] = (_Float16)w1[j]; }
      }
      *(v8h*)&Bs[srow][scol] = h;
    }
    __syncthreads();

    // A fragment: halves [kh*16, +16) of row m
    const v8h* arp = (const v8h*)&As[wm * 16 + mr][0];
    const v16h af = __builtin_shufflevector(arp[2 * kh], arp[2 * kh + 1],
        0,1,2,3,4,5,6,7,8,9,10,11,12,13,14,15);
    // B fragment: halves [kh*16, +16) of column-row n
    const v8h* b0p = (const v8h*)&Bs[wn * 32 + mr][0];
    const v8h* b1p = (const v8h*)&Bs[wn * 32 + 16 + mr][0];
    const v16h bf0 = __builtin_shufflevector(b0p[2 * kh], b0p[2 * kh + 1],
        0,1,2,3,4,5,6,7,8,9,10,11,12,13,14,15);
    const v16h bf1 = __builtin_shufflevector(b1p[2 * kh], b1p[2 * kh + 1],
        0,1,2,3,4,5,6,7,8,9,10,11,12,13,14,15);

    acc0 = __builtin_amdgcn_wmma_f32_16x16x32_f16(false, af, false, bf0,
                                                  (short)0, acc0, false, false);
    acc1 = __builtin_amdgcn_wmma_f32_16x16x32_f16(false, af, false, bf1,
                                                  (short)0, acc1, false, false);
    __syncthreads();
  }

  // epilogue: C/D layout -> row = (lane>>4)*8 + i, col = lane&15
  #pragma unroll
  for (int i = 0; i < 8; ++i) {
    const int row = mbase + wm * 16 + kh * 8 + i;
    #pragma unroll
    for (int half = 0; half < 2; ++half) {
      const int col = nbase + wn * 32 + half * 16 + mr;
      if (col < N) {
        float v = half ? acc1[i] : acc0[i];
        if (bias)   v += bias[col];
        if (addsrc) v += addsrc[(size_t)row * N + col];
        if (act == 1)      v = gelu_exact(v);
        else if (act == 2) v = tanhf(v);
        C[(size_t)row * N + col] = v;
      }
    }
  }
}

// ---------------------------------------------------------------------------
// LayerNorm over rows of length D (D<=1024): dst = (x-mean)/sqrt(var+eps)*w+b
// ---------------------------------------------------------------------------
__global__ __launch_bounds__(256)
void ln_kernel(const float* __restrict__ src, const float* __restrict__ w,
               const float* __restrict__ b, float* __restrict__ dst, int D)
{
  __shared__ float red[256];
  const int row = blockIdx.x;
  const int tid = threadIdx.x;
  float vals[4];
  int cnt = 0; float s = 0.f;
  for (int j = tid; j < D; j += 256) { float x = src[(size_t)row * D + j]; vals[cnt++] = x; s += x; }
  const float mean = block_reduce_sum(s, red) / (float)D;
  cnt = 0; s = 0.f;
  for (int j = tid; j < D; j += 256) { float d = vals[cnt++] - mean; s += d * d; }
  const float inv = rsqrtf(block_reduce_sum(s, red) / (float)D + 1e-5f);
  cnt = 0;
  for (int j = tid; j < D; j += 256)
    dst[(size_t)row * D + j] = (vals[cnt++] - mean) * inv * w[j] + b[j];
}

// ---------------------------------------------------------------------------
// Fused MHA: per (b, 16-q-row tile), loop over 16 heads.
// scores -> softmax -> P*V, accumulating mean-over-heads attention in LDS.
// qkv rows have stride 3*DM (q | k | v).
// ---------------------------------------------------------------------------
__global__ __launch_bounds__(256)
void attn_kernel(const float* __restrict__ qkv, float* __restrict__ attn_mean,
                 float* __restrict__ O)
{
  __shared__ float Ks[DL * HDIM];
  __shared__ float Vs[DL * HDIM];
  __shared__ float Qs[16 * HDIM];
  __shared__ float Ss[16 * DL];
  __shared__ float Am[16 * DL];
  __shared__ float red[256];
  __shared__ float rmax[16], rsum[16];

  const int b = blockIdx.x / QTB;
  const int qbase = (blockIdx.x % QTB) * 16;
  const int tid = threadIdx.x;

  for (int t = tid; t < 16 * DL; t += 256) Am[t] = 0.f;
  __syncthreads();

  for (int h = 0; h < NHEADS; ++h) {
    for (int t = tid; t < DL * HDIM; t += 256) {
      const int l = t / HDIM, d = t % HDIM;
      const size_t rb = (size_t)(b * DL + l) * (3 * DM);
      Ks[t] = qkv[rb + DM     + h * HDIM + d];
      Vs[t] = qkv[rb + 2 * DM + h * HDIM + d];
    }
    for (int t = tid; t < 16 * HDIM; t += 256) {
      const int r = t / HDIM, d = t % HDIM;
      const int q = qbase + r;
      Qs[t] = (q < DL) ? qkv[(size_t)(b * DL + q) * (3 * DM) + h * HDIM + d] : 0.f;
    }
    __syncthreads();

    const int rr = tid >> 4, tt = tid & 15;
    float lmax = -1e30f;
    for (int kk = tt; kk < DL; kk += 16) {
      float s = 0.f;
      #pragma unroll
      for (int d = 0; d < HDIM; ++d) s += Qs[rr * HDIM + d] * Ks[kk * HDIM + d];
      s *= 0.22360679775f;            // 1/sqrt(20)
      Ss[rr * DL + kk] = s;
      lmax = fmaxf(lmax, s);
    }
    red[tid] = lmax; __syncthreads();
    if (tt == 0) {
      float m = red[rr * 16];
      for (int j = 1; j < 16; ++j) m = fmaxf(m, red[rr * 16 + j]);
      rmax[rr] = m;
    }
    __syncthreads();
    const float m = rmax[rr];
    float lsum = 0.f;
    for (int kk = tt; kk < DL; kk += 16) {
      float e = expf(Ss[rr * DL + kk] - m);
      Ss[rr * DL + kk] = e; lsum += e;
    }
    red[tid] = lsum; __syncthreads();
    if (tt == 0) {
      float s2 = 0.f;
      for (int j = 0; j < 16; ++j) s2 += red[rr * 16 + j];
      rsum[rr] = s2;
    }
    __syncthreads();
    const float inv = 1.0f / rsum[rr];
    for (int kk = tt; kk < DL; kk += 16) {
      const float pv = Ss[rr * DL + kk] * inv;
      Ss[rr * DL + kk] = pv;
      Am[rr * DL + kk] += pv * (1.0f / NHEADS);
    }
    __syncthreads();
    for (int t = tid; t < 16 * HDIM; t += 256) {
      const int r = t / HDIM, d = t % HDIM;
      const int q = qbase + r;
      if (q < DL) {
        float acc = 0.f;
        for (int kk = 0; kk < DL; ++kk) acc += Ss[r * DL + kk] * Vs[kk * HDIM + d];
        O[(size_t)(b * DL + q) * DM + h * HDIM + d] = acc;
      }
    }
    __syncthreads();
  }
  for (int t = tid; t < 16 * DL; t += 256) {
    const int r = t / DL, kk = t % DL;
    const int q = qbase + r;
    if (q < DL) attn_mean[((size_t)b * DL + q) * DL + kk] = Am[t];
  }
}

// ---------------------------------------------------------------------------
// x (B,C,L) -> xT (B,L,KC) zero-padded in channel dim (KC=160, K%32==0)
// ---------------------------------------------------------------------------
__global__ void transpose_kernel(const float* __restrict__ x, float* __restrict__ xT)
{
  const size_t idx = (size_t)blockIdx.x * 256 + threadIdx.x;
  if (idx >= (size_t)DROWS * KC) return;
  const int c = (int)(idx % KC);
  const size_t row = idx / KC;
  const int l = (int)(row % DL);
  const int b = (int)(row / DL);
  xT[idx] = (c < DC) ? x[((size_t)b * DC + c) * DL + l] : 0.f;
}

// sp_conv_w (320,129) -> (320,160) zero-padded
__global__ void padw_kernel(const float* __restrict__ w, float* __restrict__ wp)
{
  const int idx = blockIdx.x * 256 + threadIdx.x;
  if (idx >= DM * KC) return;
  const int c = idx % KC, n = idx / KC;
  wp[idx] = (c < DC) ? w[n * DC + c] : 0.f;
}

// dt = softplus(zx[...,1344+h] + dt_bias), dA = exp(-exp(A_log)*dt)
__global__ void dt_kernel(const float* __restrict__ zx, const float* __restrict__ dtb,
                          const float* __restrict__ alog,
                          float* __restrict__ dtv, float* __restrict__ dav)
{
  const size_t idx = (size_t)blockIdx.x * 256 + threadIdx.x;
  if (idx >= (size_t)DROWS * HM) return;
  const int h = (int)(idx % HM);
  const size_t row = idx / HM;
  const float raw = zx[row * DPROJ + 1344 + h] + dtb[h];
  const float sp = (raw > 30.f) ? raw : log1pf(expf(raw));
  dtv[idx] = sp;
  dav[idx] = expf(-expf(alog[h]) * sp);
}

// depthwise causal conv (D_CONV=4) over xBC = zx cols [640,1344), then SiLU
__global__ void conv_silu_kernel(const float* __restrict__ zx, const float* __restrict__ cw,
                                 const float* __restrict__ cb, float* __restrict__ xc)
{
  const size_t idx = (size_t)blockIdx.x * 256 + threadIdx.x;
  if (idx >= (size_t)DROWS * CDIM) return;
  const int ch = (int)(idx % CDIM);
  const size_t row = idx / CDIM;
  const int l = (int)(row % DL);
  const size_t brow = row - l;         // b*DL
  float acc = cb[ch];
  #pragma unroll
  for (int k = 0; k < 4; ++k) {
    const int ll = l + k - 3;
    if (ll >= 0) acc += zx[(brow + ll) * DPROJ + 640 + ch] * cw[ch * 4 + k];
  }
  xc[idx] = acc * sigm(acc);
}

// ---------------------------------------------------------------------------
// SSM scan: block per (b,h); thread owns state[p, nb..nb+3] in registers.
// Per step l: broadcast B/C/x/dA/dt via LDS, update, reduce over n, emit y.
// ---------------------------------------------------------------------------
__global__ __launch_bounds__(256)
void scan_kernel(const float* __restrict__ dav, const float* __restrict__ dtv,
                 const float* __restrict__ xc, const float* __restrict__ Dsk,
                 float* __restrict__ y)
{
  __shared__ float Bsh[32], Csh[32], Xsh[32], part[256], sc[2];
  const int b = blockIdx.x / HM, h = blockIdx.x % HM;
  const int tid = threadIdx.x;
  const int p = tid >> 3, nb = (tid & 7) * 4;
  float st0 = 0.f, st1 = 0.f, st2 = 0.f, st3 = 0.f;
  const float Dh = Dsk[h];

  for (int l = 0; l < DL; ++l) {
    const size_t base = (size_t)(b * DL + l);
    if (tid < 32)      { Bsh[tid] = xc[base * CDIM + 640 + tid];
                         Csh[tid] = xc[base * CDIM + 672 + tid]; }
    else if (tid < 64) { const int pp = tid - 32;
                         Xsh[pp] = xc[base * CDIM + h * 32 + pp]; }
    else if (tid == 64){ sc[0] = dav[base * HM + h];
                         sc[1] = dtv[base * HM + h]; }
    __syncthreads();
    const float da = sc[0];
    const float dtx = sc[1] * Xsh[p];
    st0 = st0 * da + dtx * Bsh[nb + 0];
    st1 = st1 * da + dtx * Bsh[nb + 1];
    st2 = st2 * da + dtx * Bsh[nb + 2];
    st3 = st3 * da + dtx * Bsh[nb + 3];
    part[tid] = st0 * Csh[nb + 0] + st1 * Csh[nb + 1] + st2 * Csh[nb + 2] + st3 * Csh[nb + 3];
    __syncthreads();
    if ((tid & 7) == 0) {
      float yv = Dh * Xsh[p];
      #pragma unroll
      for (int j = 0; j < 8; ++j) yv += part[(p << 3) + j];
      y[base * DIN + h * 32 + p] = yv;
    }
    __syncthreads();
  }
}

// g = y * z*sigmoid(z); g *= rsqrt(mean(g^2)+eps) * gn_w  (row of 640, in-place y)
__global__ __launch_bounds__(256)
void gate_rms_kernel(const float* __restrict__ zx, float* __restrict__ y,
                     const float* __restrict__ gw)
{
  __shared__ float red[256];
  const int row = blockIdx.x;
  const int tid = threadIdx.x;
  float vals[3];
  int cnt = 0; float s = 0.f;
  for (int j = tid; j < DIN; j += 256) {
    const float z  = zx[(size_t)row * DPROJ + j];
    const float yv = y[(size_t)row * DIN + j];
    const float g  = yv * (z * sigm(z));
    vals[cnt++] = g; s += g * g;
  }
  const float scale = rsqrtf(block_reduce_sum(s, red) / (float)DIN + 1e-5f);
  cnt = 0;
  for (int j = tid; j < DIN; j += 256)
    y[(size_t)row * DIN + j] = vals[cnt++] * scale * gw[j];
}

// attention pooling: s[l]=hr80[l]·w2+b2; softmax over L; a out; pooled = sum a*h
__global__ __launch_bounds__(256)
void pool_kernel(const float* __restrict__ h, const float* __restrict__ hr80,
                 const float* __restrict__ w2, const float* __restrict__ b2,
                 float* __restrict__ aout, float* __restrict__ pooled)
{
  __shared__ float s[DL];
  __shared__ float red[256];
  const int b = blockIdx.x, tid = threadIdx.x;
  if (tid < DL) {
    float acc = b2[0];
    const float* hp = hr80 + (size_t)(b * DL + tid) * 80;
    for (int j = 0; j < 80; ++j) acc += hp[j] * w2[j];
    s[tid] = acc;
  }
  __syncthreads();
  const float mx  = block_reduce_max((tid < DL) ? s[tid] : -1e30f, red);
  const float e   = (tid < DL) ? expf(s[tid] - mx) : 0.f;
  const float tot = block_reduce_sum(e, red);
  if (tid < DL) {
    const float pr = e / tot;
    s[tid] = pr;
    aout[(size_t)b * DL + tid] = pr;
  }
  __syncthreads();
  for (int d = tid; d < DM; d += 256) {
    float acc = 0.f;
    for (int l = 0; l < DL; ++l) acc += h[(size_t)(b * DL + l) * DM + d] * s[l];
    pooled[(size_t)b * DM + d] = acc;
  }
}

// head: hr = gelu(pooled·w1^T + b1) (160); out[b] = hr·w2 + b2
__global__ __launch_bounds__(256)
void head_kernel(const float* __restrict__ pooled, const float* __restrict__ w1,
                 const float* __restrict__ b1, const float* __restrict__ w2,
                 const float* __restrict__ b2, float* __restrict__ out)
{
  __shared__ float hr[160];
  __shared__ float red[256];
  const int b = blockIdx.x, tid = threadIdx.x;
  if (tid < 160) {
    float acc = b1[tid];
    const float* pp = pooled + (size_t)b * DM;
    const float* wp = w1 + (size_t)tid * DM;
    for (int k = 0; k < DM; ++k) acc += pp[k] * wp[k];
    hr[tid] = gelu_exact(acc);
  }
  __syncthreads();
  const float tot = block_reduce_sum((tid < 160) ? hr[tid] * w2[tid] : 0.f, red);
  if (tid == 0) out[b] = tot + b2[0];
}

// ---------------------------------------------------------------------------
extern "C" void kernel_launch(void* const* d_in, const int* in_sizes, int n_in,
                              void* d_out, int out_size, void* d_ws, size_t ws_size,
                              hipStream_t stream)
{
  (void)in_sizes; (void)n_in; (void)out_size; (void)ws_size;

  const float* x         = (const float*)d_in[0];
  const float* sp_conv_w = (const float*)d_in[1];
  const float* sp_conv_b = (const float*)d_in[2];
  const float* sp_qkv_w  = (const float*)d_in[3];
  const float* sp_qkv_b  = (const float*)d_in[4];
  const float* sp_out_w  = (const float*)d_in[5];
  const float* sp_out_b  = (const float*)d_in[6];
  const float* sp_ln1_w  = (const float*)d_in[7];
  const float* sp_ln1_b  = (const float*)d_in[8];
  const float* sp_ff1_w  = (const float*)d_in[9];
  const float* sp_ff1_b  = (const float*)d_in[10];
  const float* sp_ff2_w  = (const float*)d_in[11];
  const float* sp_ff2_b  = (const float*)d_in[12];
  const float* sp_ln2_w  = (const float*)d_in[13];
  const float* sp_ln2_b  = (const float*)d_in[14];
  const float* norm_in_w = (const float*)d_in[15];
  const float* norm_in_b = (const float*)d_in[16];
  const float* m_in_w    = (const float*)d_in[17];
  const float* m_conv_w  = (const float*)d_in[18];
  const float* m_conv_b  = (const float*)d_in[19];
  const float* m_dt_bias = (const float*)d_in[20];
  const float* m_A_log   = (const float*)d_in[21];
  const float* m_D       = (const float*)d_in[22];
  const float* m_gnorm_w = (const float*)d_in[23];
  const float* m_out_w   = (const float*)d_in[24];
  const float* ln_w      = (const float*)d_in[25];
  const float* ln_b      = (const float*)d_in[26];
  const float* pm_qkv_w  = (const float*)d_in[27];
  const float* pm_qkv_b  = (const float*)d_in[28];
  const float* pm_out_w  = (const float*)d_in[29];
  const float* pm_out_b  = (const float*)d_in[30];
  const float* pm_ln_w   = (const float*)d_in[31];
  const float* pm_ln_b   = (const float*)d_in[32];
  const float* pool_w1   = (const float*)d_in[33];
  const float* pool_b1   = (const float*)d_in[34];
  const float* pool_w2   = (const float*)d_in[35];
  const float* pool_b2   = (const float*)d_in[36];
  const float* head_w1   = (const float*)d_in[37];
  const float* head_b1   = (const float*)d_in[38];
  const float* head_w2   = (const float*)d_in[39];
  const float* head_b2   = (const float*)d_in[40];

  const size_t M = DROWS;
  float* ws  = (float*)d_ws;
  float* H   = ws;                           // (M,320) hidden
  float* R   = H   + (size_t)320 * M;        // (M,320) pre-LN sum
  float* LNB = R   + (size_t)320 * M;        // (M,320) LN(h) for mamba
  float* ZX  = LNB + (size_t)320 * M;        // (M,1364) zxbcdt / qkv(960) / ff-mid(640)
  float* XC  = ZX  + (size_t)1364 * M;       // (M,704) conv out; alias: attn O (M,320)
  float* Y   = XC  + (size_t)704 * M;        // (M,640) scan out; alias: xT (M,160)
  float* DTB = Y   + (size_t)640 * M;        // (M,20)
  float* DAB = DTB + (size_t)20 * M;         // (M,20)
  float* PO  = DAB + (size_t)20 * M;         // (128,320) pooled
  float* WPAD = PO + (size_t)DB * DM;        // (320,160) padded sp_conv_w
  float* XT   = Y;
  float* OBUF = XC;
  float* HR80 = ZX;

  float* out0   = (float*)d_out;                          // (128,)
  float* out_sp = out0 + DB;                              // (128,200,200)
  float* out_pm = out_sp + (size_t)DB * DL * DL;          // (128,200,200)
  float* out_a  = out_pm + (size_t)DB * DL * DL;          // (128,200)

  const dim3 blk(256);
  auto gemm = [&](const float* A, const float* W, const float* bias, const float* add,
                  float* C, int N, int K, int act) {
    dim3 grid((unsigned)((N + 63) / 64), (unsigned)(M / 64));
    gemm_wmma_kernel<<<grid, blk, 0, stream>>>(A, W, bias, add, C, (int)M, N, K, act);
  };
  auto mha = [&](const float* qkvw, const float* qkvb, const float* ow, const float* ob,
                 const float* lnw, const float* lnb, float* attn_out) {
    gemm(H, qkvw, qkvb, nullptr, ZX, 3 * DM, DM, 0);                 // qkv (stride 960)
    attn_kernel<<<DB * QTB, blk, 0, stream>>>(ZX, attn_out, OBUF);
    gemm(OBUF, ow, ob, H, R, DM, DM, 0);                             // h + out-proj
    ln_kernel<<<(unsigned)M, blk, 0, stream>>>(R, lnw, lnb, H, DM);
  };

  // --- spatial conv as GEMM (K padded 129->160) ---
  { const size_t n = (size_t)DROWS * KC;
    transpose_kernel<<<(unsigned)((n + 255) / 256), blk, 0, stream>>>(x, XT); }
  padw_kernel<<<(DM * KC + 255) / 256, blk, 0, stream>>>(sp_conv_w, WPAD);
  gemm(XT, WPAD, sp_conv_b, nullptr, H, DM, KC, 0);

  // --- spatial transformer block ---
  mha(sp_qkv_w, sp_qkv_b, sp_out_w, sp_out_b, sp_ln1_w, sp_ln1_b, out_sp);
  gemm(H, sp_ff1_w, sp_ff1_b, nullptr, ZX, 2 * DM, DM, 1);           // gelu
  gemm(ZX, sp_ff2_w, sp_ff2_b, H, R, DM, 2 * DM, 0);
  ln_kernel<<<(unsigned)M, blk, 0, stream>>>(R, sp_ln2_w, sp_ln2_b, H, DM);
  ln_kernel<<<(unsigned)M, blk, 0, stream>>>(H, norm_in_w, norm_in_b, H, DM);

  // --- 4 mamba2 layers ---
  for (int i = 0; i < 4; ++i) {
    ln_kernel<<<(unsigned)M, blk, 0, stream>>>(H, ln_w + (size_t)i * DM,
                                               ln_b + (size_t)i * DM, LNB, DM);
    gemm(LNB, m_in_w + (size_t)i * DPROJ * DM, nullptr, nullptr, ZX, DPROJ, DM, 0);
    { const size_t n = (size_t)M * HM;
      dt_kernel<<<(unsigned)((n + 255) / 256), blk, 0, stream>>>(
          ZX, m_dt_bias + (size_t)i * HM, m_A_log + (size_t)i * HM, DTB, DAB); }
    { const size_t n = (size_t)M * CDIM;
      conv_silu_kernel<<<(unsigned)((n + 255) / 256), blk, 0, stream>>>(
          ZX, m_conv_w + (size_t)i * CDIM * 4, m_conv_b + (size_t)i * CDIM, XC); }
    scan_kernel<<<DB * HM, blk, 0, stream>>>(DAB, DTB, XC, m_D + (size_t)i * HM, Y);
    gate_rms_kernel<<<(unsigned)M, blk, 0, stream>>>(ZX, Y, m_gnorm_w + (size_t)i * DIN);
    gemm(Y, m_out_w + (size_t)i * DM * DIN, nullptr, H, H, DM, DIN, 0);  // + residual
  }

  // --- post transformer block ---
  mha(pm_qkv_w, pm_qkv_b, pm_out_w, pm_out_b, pm_ln_w, pm_ln_b, out_pm);

  // --- attention pooling + head ---
  gemm(H, pool_w1, pool_b1, nullptr, HR80, 80, DM, 2);               // tanh
  pool_kernel<<<DB, blk, 0, stream>>>(H, HR80, pool_w2, pool_b2, out_a, PO);
  head_kernel<<<DB, blk, 0, stream>>>(PO, head_w1, head_b1, head_w2, head_b2, out0);
}
